// Atten_52450140619066
// MI455X (gfx1250) — matrix operand order repaired and checked by reading, not compiled, not run on hardware
//
#include <hip/hip_runtime.h>
#include <hip/hip_bf16.h>

#define BB   8
#define CC   128
#define HH   128
#define WW2  128
#define HWN  16384

typedef __attribute__((ext_vector_type(16))) _Float16 v16h;
typedef __attribute__((ext_vector_type(8)))  float    v8f;
typedef __attribute__((ext_vector_type(8)))  _Float16 h8;
typedef __attribute__((ext_vector_type(2)))  _Float16 h2;

__device__ __forceinline__ float wave_sum(float v) {
#pragma unroll
  for (int off = 16; off > 0; off >>= 1) v += __shfl_xor(v, off, 32);
  return v;
}
__device__ __forceinline__ float wave_max(float v) {
#pragma unroll
  for (int off = 16; off > 0; off >>= 1) v = fmaxf(v, __shfl_xor(v, off, 32));
  return v;
}

// ---------------------------------------------------------------------------
// K1: fused channel-LayerNorm + conv1x1 (GEMM over C=128) -> f16 [B,OC,HW]
// block: 256 thr (8 waves); tile M=64 positions x N=128 out-channels
// ---------------------------------------------------------------------------
__global__ __launch_bounds__(256) void ln_gemm_kernel(
    const float* __restrict__ x,     // [B,HW,C]
    const float* __restrict__ lnw, const float* __restrict__ lnb,
    const float* __restrict__ w1,    // [OC,C]
    const float* __restrict__ b1,    // [OC]
    _Float16* __restrict__ y,        // [B,OC,HW]
    int OC)
{
  __shared__ _Float16 Xs[64][132];
  __shared__ _Float16 Ws[128][132];

  const int tid  = threadIdx.x;
  const int wave = tid >> 5;
  const int lane = tid & 31;
  const int lm = lane & 15, lh = lane >> 4;
  const int p0  = blockIdx.x * 64;
  const int oc0 = blockIdx.y * 128;
  const int b   = blockIdx.z;

  const float4 gw = *(const float4*)(lnw + lane * 4);
  const float4 gb = *(const float4*)(lnb + lane * 4);

  // LayerNorm over C for 64 positions -> Xs (one row per wave per iter)
#pragma unroll
  for (int it = 0; it < 8; ++it) {
    const int row = wave + it * 8;
    float4 v = *(const float4*)(x + ((size_t)b * HWN + p0 + row) * CC + lane * 4);
    float s  = wave_sum(v.x + v.y + v.z + v.w);
    float sq = wave_sum(v.x * v.x + v.y * v.y + v.z * v.z + v.w * v.w);
    float mu   = s * (1.0f / CC);
    float rstd = rsqrtf(sq * (1.0f / CC) - mu * mu + 1e-5f);
    Xs[row][lane * 4 + 0] = (_Float16)((v.x - mu) * rstd * gw.x + gb.x);
    Xs[row][lane * 4 + 1] = (_Float16)((v.y - mu) * rstd * gw.y + gb.y);
    Xs[row][lane * 4 + 2] = (_Float16)((v.z - mu) * rstd * gw.z + gb.z);
    Xs[row][lane * 4 + 3] = (_Float16)((v.w - mu) * rstd * gw.w + gb.w);
  }
  // weight slice [128 out-ch x 128 in-ch] -> Ws
  for (int i = tid; i < 128 * 32; i += 256) {
    const int n  = i >> 5;
    const int c4 = (i & 31) * 4;
    float4 wv = *(const float4*)(w1 + (size_t)(oc0 + n) * CC + c4);
    Ws[n][c4 + 0] = (_Float16)wv.x;
    Ws[n][c4 + 1] = (_Float16)wv.y;
    Ws[n][c4 + 2] = (_Float16)wv.z;
    Ws[n][c4 + 3] = (_Float16)wv.w;
  }
  __syncthreads();

  const int m0 = (wave & 3) * 16;
  const int ng = (wave >> 2) * 64;

  // hoist A fragments (depend only on K-chunk, shared by all 4 N-tiles)
  v16h afrag[4];
#pragma unroll
  for (int kk = 0; kk < 4; ++kk) {
    const int k0 = kk * 32;
#pragma unroll
    for (int v = 0; v < 8; ++v) {
      const int ka = k0 + (v >> 2) * 16 + lh * 8 + (v & 3) * 2;   // A 16x32 layout
      afrag[kk][2 * v + 0] = Xs[m0 + lm][ka + 0];
      afrag[kk][2 * v + 1] = Xs[m0 + lm][ka + 1];
    }
  }

#pragma unroll
  for (int nt = 0; nt < 4; ++nt) {
    const int n0 = ng + nt * 16;
    v8f acc = {0.f, 0.f, 0.f, 0.f, 0.f, 0.f, 0.f, 0.f};
#pragma unroll
    for (int kk = 0; kk < 4; ++kk) {
      const int k0 = kk * 32;
      v16h bm;
#pragma unroll
      for (int v = 0; v < 8; ++v) {
        const int kb = k0 + (v >> 2) * 16 + lh * 8 + (v & 3) * 2; // B 32x16 layout
        bm[2 * v + 0] = Ws[n0 + lm][kb + 0];
        bm[2 * v + 1] = Ws[n0 + lm][kb + 1];
      }
      acc = __builtin_amdgcn_wmma_f32_16x16x32_f16(false, afrag[kk], false, bm,
                                                   (short)0, acc, false, false);
    }
    // D layout: lane owns (N=lm, M=lh*8 + r) -> 8 consecutive M: vector store
    const float bias = b1[oc0 + n0 + lm];
    h8 ov;
#pragma unroll
    for (int r = 0; r < 8; ++r) ov[r] = (_Float16)(acc[r] + bias);
    *(h8*)(y + ((size_t)b * OC + oc0 + n0 + lm) * HWN + p0 + m0 + lh * 8) = ov;
  }
}

// ---------------------------------------------------------------------------
// K2: depthwise 3x3 conv + bias, f16 in/out, f32 accumulate
// ---------------------------------------------------------------------------
__global__ __launch_bounds__(256) void dwconv_kernel(
    const _Float16* __restrict__ in, const float* __restrict__ wdw,
    const float* __restrict__ bdw, _Float16* __restrict__ out, int OC)
{
  const size_t gid = (size_t)blockIdx.x * 256 + threadIdx.x;
  const int xx = (int)(gid & (WW2 - 1));
  const int yy = (int)((gid >> 7) & (HH - 1));
  const size_t chb = gid >> 14;                 // b*OC + ch
  const int ch = (int)(chb % (size_t)OC);
  const float* wp = wdw + ch * 9;
  const _Float16* ip = in + (chb << 14);
  float s = bdw[ch];
#pragma unroll
  for (int dy = -1; dy <= 1; ++dy) {
    const int y2 = yy + dy;
    if (y2 < 0 || y2 >= HH) continue;
#pragma unroll
    for (int dx = -1; dx <= 1; ++dx) {
      const int x2 = xx + dx;
      if (x2 < 0 || x2 >= WW2) continue;
      s += wp[(dy + 1) * 3 + (dx + 1)] * (float)ip[y2 * WW2 + x2];
    }
  }
  out[gid] = (_Float16)s;
}

// ---------------------------------------------------------------------------
// K3: reciprocal L2 norms over HW for q (z=0) and k (z=1)
// ---------------------------------------------------------------------------
__global__ __launch_bounds__(256) void l2norm_kernel(
    const _Float16* __restrict__ q2, const _Float16* __restrict__ kv2,
    float* __restrict__ rq, float* __restrict__ rk)
{
  __shared__ float red[8];
  const int tid = threadIdx.x, wave = tid >> 5, lane = tid & 31;
  const int c = blockIdx.x, b = blockIdx.y, which = blockIdx.z;
  const _Float16* src = (which == 0)
      ? (q2  + ((size_t)b * CC + c) * HWN)
      : (kv2 + ((size_t)b * 2 * CC + c) * HWN);
  float s = 0.f;
  for (int i = tid * 8; i < HWN; i += 256 * 8) {
    h8 v = *(const h8*)(src + i);
#pragma unroll
    for (int j = 0; j < 8; ++j) { float f = (float)v[j]; s += f * f; }
  }
  s = wave_sum(s);
  if (lane == 0) red[wave] = s;
  __syncthreads();
  if (tid == 0) {
    float t = 0.f;
#pragma unroll
    for (int wv = 0; wv < 8; ++wv) t += red[wv];
    float* dst = which ? rk : rq;
    dst[b * CC + c] = 1.0f / fmaxf(sqrtf(t), 1e-12f);
  }
}

__global__ void zero_kernel(float* __restrict__ p, int n) {
  int i = blockIdx.x * 256 + threadIdx.x;
  if (i < n) p[i] = 0.f;
}

// ---------------------------------------------------------------------------
// K4: split-K q·k^T  (M=N=128 channels, K=HW), atomic f32 accumulate
// grid: (HW/128, B); block 256 thr; wave w owns M-tile w, loops all 8 N-tiles
// ---------------------------------------------------------------------------
__global__ __launch_bounds__(256) void qk_gemm_kernel(
    const _Float16* __restrict__ q2, const _Float16* __restrict__ kv2,
    float* __restrict__ att_raw)
{
  const int tid = threadIdx.x, wave = tid >> 5, lane = tid & 31;
  const int lm = lane & 15, lh = lane >> 4;
  const int p0 = blockIdx.x * 128;
  const int b  = blockIdx.y;
  const int m0 = wave * 16;
  const _Float16* qrow = q2 + ((size_t)b * CC + m0 + lm) * HWN + p0;

  // hoist A fragments (q row slice), reused by all 8 N-tiles
  v16h afrag[4];
#pragma unroll
  for (int kk = 0; kk < 4; ++kk) {
    const int k0 = kk * 32;
#pragma unroll
    for (int v = 0; v < 8; ++v) {
      const int ka = k0 + (v >> 2) * 16 + lh * 8 + (v & 3) * 2;
      h2 ta = *(const h2*)(qrow + ka);
      afrag[kk][2 * v + 0] = ta[0];
      afrag[kk][2 * v + 1] = ta[1];
    }
  }

#pragma unroll
  for (int nt = 0; nt < 8; ++nt) {
    const int n0 = nt * 16;
    const _Float16* krow = kv2 + ((size_t)b * 2 * CC + n0 + lm) * HWN + p0;
    v8f acc = {0.f, 0.f, 0.f, 0.f, 0.f, 0.f, 0.f, 0.f};
#pragma unroll
    for (int kk = 0; kk < 4; ++kk) {
      const int k0 = kk * 32;
      v16h bm;
#pragma unroll
      for (int v = 0; v < 8; ++v) {
        const int kb = k0 + (v >> 2) * 16 + lh * 8 + (v & 3) * 2;
        h2 tb = *(const h2*)(krow + kb);
        bm[2 * v + 0] = tb[0];
        bm[2 * v + 1] = tb[1];
      }
      acc = __builtin_amdgcn_wmma_f32_16x16x32_f16(false, afrag[kk], false, bm,
                                                   (short)0, acc, false, false);
    }
#pragma unroll
    for (int r = 0; r < 8; ++r)
      atomicAdd(att_raw + ((size_t)b * CC + m0 + r + 8 * lh) * CC + n0 + lm, acc[r]);
  }
}

// ---------------------------------------------------------------------------
// K5: softmax over last dim with rq*rk scaling; emits f16 attention
// ---------------------------------------------------------------------------
__global__ __launch_bounds__(128) void softmax_kernel(
    const float* __restrict__ att_raw, const float* __restrict__ rq,
    const float* __restrict__ rk, _Float16* __restrict__ attH)
{
  __shared__ float red[4];
  const int i = blockIdx.x, b = blockIdx.y;
  const int j = threadIdx.x;
  const int wave = j >> 5, lane = j & 31;
  float v = att_raw[((size_t)b * CC + i) * CC + j] * rq[b * CC + i] * rk[b * CC + j];
  float m = wave_max(v);
  if (lane == 0) red[wave] = m;
  __syncthreads();
  m = fmaxf(fmaxf(red[0], red[1]), fmaxf(red[2], red[3]));
  __syncthreads();
  float e = __expf(v - m);
  float s = wave_sum(e);
  if (lane == 0) red[wave] = s;
  __syncthreads();
  s = red[0] + red[1] + red[2] + red[3];
  attH[((size_t)b * CC + i) * CC + j] = (_Float16)(e / s);
}

// ---------------------------------------------------------------------------
// K6: fused  out = (att @ v) -> conv1x1(out_w) + out_b + cur, store [B,HW,C]
// grid: (HW/128, B); block 256 thr.  GEMM1: S[c][p] = att@v (S kept in LDS),
// GEMM2: y[p][oc] = S^T @ out_w^T, residual + transpose via per-wave LDS tile.
// ---------------------------------------------------------------------------
__global__ __launch_bounds__(256) void av_out_kernel(
    const _Float16* __restrict__ attH, const _Float16* __restrict__ kv2,
    const float* __restrict__ ow, const float* __restrict__ ob,
    const float* __restrict__ cur, float* __restrict__ out)
{
  __shared__ _Float16 Vs[128][136];   // V tile, later reused as S[c][p]
  __shared__ float Ds[8][16][20];     // per-wave 16x16 transpose stage
  const int tid = threadIdx.x, wave = tid >> 5, lane = tid & 31;
  const int lm = lane & 15, lh = lane >> 4;
  const int p0 = blockIdx.x * 128;
  const int b  = blockIdx.y;

  // stage V chunk: channels d=0..127 live at kv rows b*2C + C + d
  for (int i = tid; i < 128 * 16; i += 256) {
    const int d = i >> 4, c8 = (i & 15) * 8;
    *(h8*)&Vs[d][c8] =
        *(const h8*)(kv2 + ((size_t)b * 2 * CC + CC + d) * HWN + p0 + c8);
  }
  __syncthreads();

  // GEMM1: S[c][p] = sum_d att[c][d] * V[d][p]
  const int m0 = wave * 16;
  v8f acc[8];
#pragma unroll
  for (int t = 0; t < 8; ++t) {
    v8f z = {0.f, 0.f, 0.f, 0.f, 0.f, 0.f, 0.f, 0.f};
    acc[t] = z;
  }
  const _Float16* arow = attH + ((size_t)b * CC + m0 + lm) * CC;

  // hoist attention A fragments (reused by all 8 N-tiles)
  v16h afrag[4];
#pragma unroll
  for (int kk = 0; kk < 4; ++kk) {
    const int k0 = kk * 32;
#pragma unroll
    for (int v = 0; v < 8; ++v) {
      const int ka = k0 + (v >> 2) * 16 + lh * 8 + (v & 3) * 2;
      h2 ta = *(const h2*)(arow + ka);
      afrag[kk][2 * v + 0] = ta[0];
      afrag[kk][2 * v + 1] = ta[1];
    }
  }

#pragma unroll
  for (int nt = 0; nt < 8; ++nt) {
    const int n0 = nt * 16;
#pragma unroll
    for (int kk = 0; kk < 4; ++kk) {
      const int k0 = kk * 32;
      v16h bm;
#pragma unroll
      for (int v = 0; v < 8; ++v) {
        const int kb = k0 + (v >> 2) * 16 + lh * 8 + (v & 3) * 2;
        bm[2 * v + 0] = Vs[kb + 0][n0 + lm];
        bm[2 * v + 1] = Vs[kb + 1][n0 + lm];
      }
      acc[nt] = __builtin_amdgcn_wmma_f32_16x16x32_f16(false, afrag[kk], false, bm,
                                                       (short)0, acc[nt], false, false);
    }
  }
  __syncthreads();                      // everyone done reading V
#pragma unroll
  for (int nt = 0; nt < 8; ++nt) {
#pragma unroll
    for (int r = 0; r < 8; ++r)
      Vs[m0 + r + 8 * lh][nt * 16 + lm] = (_Float16)acc[nt][r];   // S[c][p]
  }
  __syncthreads();

  // GEMM2: y[p][oc] = sum_c S[c][p] * out_w[oc][c]  (+bias +residual)
  // hoist S A fragments (depend only on K-chunk)
  v16h afrag2[4];
#pragma unroll
  for (int kk = 0; kk < 4; ++kk) {
    const int k0 = kk * 32;
#pragma unroll
    for (int v = 0; v < 8; ++v) {
      const int ka = k0 + (v >> 2) * 16 + lh * 8 + (v & 3) * 2;
      afrag2[kk][2 * v + 0] = Vs[ka + 0][m0 + lm];
      afrag2[kk][2 * v + 1] = Vs[ka + 1][m0 + lm];
    }
  }

#pragma unroll
  for (int nt = 0; nt < 8; ++nt) {
    const int n0 = nt * 16;
    const float* wrow = ow + (size_t)(n0 + lm) * CC;
    v8f a2 = {0.f, 0.f, 0.f, 0.f, 0.f, 0.f, 0.f, 0.f};
#pragma unroll
    for (int kk = 0; kk < 4; ++kk) {
      const int k0 = kk * 32;
      v16h bm;
#pragma unroll
      for (int v = 0; v < 8; ++v) {
        const int kb = k0 + (v >> 2) * 16 + lh * 8 + (v & 3) * 2;
        bm[2 * v + 0] = (_Float16)wrow[kb + 0];
        bm[2 * v + 1] = (_Float16)wrow[kb + 1];
      }
      a2 = __builtin_amdgcn_wmma_f32_16x16x32_f16(false, afrag2[kk], false, bm,
                                                  (short)0, a2, false, false);
    }
    const float bias = ob[n0 + lm];
#pragma unroll
    for (int r = 0; r < 8; ++r)
      Ds[wave][r + 8 * lh][lm] = a2[r] + bias;   // [m=p_local][n=oc_local]
    // transposed read: lane owns row p = lm, consecutive oc -> float4 stores
    const size_t gp = (size_t)b * HWN + p0 + m0 + lm;
    float4 c0 = *(const float4*)(cur + gp * CC + n0 + lh * 8);
    float4 c1 = *(const float4*)(cur + gp * CC + n0 + lh * 8 + 4);
    float4 o0, o1;
    o0.x = Ds[wave][lm][lh * 8 + 0] + c0.x;
    o0.y = Ds[wave][lm][lh * 8 + 1] + c0.y;
    o0.z = Ds[wave][lm][lh * 8 + 2] + c0.z;
    o0.w = Ds[wave][lm][lh * 8 + 3] + c0.w;
    o1.x = Ds[wave][lm][lh * 8 + 4] + c1.x;
    o1.y = Ds[wave][lm][lh * 8 + 5] + c1.y;
    o1.z = Ds[wave][lm][lh * 8 + 6] + c1.z;
    o1.w = Ds[wave][lm][lh * 8 + 7] + c1.w;
    *(float4*)(out + gp * CC + n0 + lh * 8)     = o0;
    *(float4*)(out + gp * CC + n0 + lh * 8 + 4) = o1;
  }
}

// ---------------------------------------------------------------------------
extern "C" void kernel_launch(void* const* d_in, const int* in_sizes, int n_in,
                              void* d_out, int out_size, void* d_ws, size_t ws_size,
                              hipStream_t stream)
{
  (void)in_sizes; (void)n_in; (void)out_size; (void)ws_size;
  const float* pre  = (const float*)d_in[0];
  const float* cur  = (const float*)d_in[1];
  const float* ln1w = (const float*)d_in[2];
  const float* ln1b = (const float*)d_in[3];
  const float* ln2w = (const float*)d_in[4];
  const float* ln2b = (const float*)d_in[5];
  const float* qw1  = (const float*)d_in[6];
  const float* qb1  = (const float*)d_in[7];
  const float* qw2  = (const float*)d_in[8];
  const float* qb2  = (const float*)d_in[9];
  const float* kvw1 = (const float*)d_in[10];
  const float* kvb1 = (const float*)d_in[11];
  const float* kvw2 = (const float*)d_in[12];
  const float* kvb2 = (const float*)d_in[13];
  const float* ow   = (const float*)d_in[14];
  const float* ob   = (const float*)d_in[15];
  float* out = (float*)d_out;

  char* ws = (char*)d_ws;
  size_t off = 0;
  auto alloc = [&](size_t bytes) -> char* {
    char* p = ws + off;
    off += (bytes + 255) & ~(size_t)255;
    return p;
  };
  _Float16* q1   = (_Float16*)alloc((size_t)BB * CC * HWN * 2);
  _Float16* kv1  = (_Float16*)alloc((size_t)BB * 2 * CC * HWN * 2);
  _Float16* q2   = (_Float16*)alloc((size_t)BB * CC * HWN * 2);
  _Float16* kv2  = (_Float16*)alloc((size_t)BB * 2 * CC * HWN * 2);
  float*    att  = (float*)   alloc((size_t)BB * CC * CC * 4);
  _Float16* attH = (_Float16*)alloc((size_t)BB * CC * CC * 2);
  float*    rq   = (float*)   alloc((size_t)BB * CC * 4);
  float*    rk   = (float*)   alloc((size_t)BB * CC * 4);

  zero_kernel<<<dim3((BB * CC * CC + 255) / 256), 256, 0, stream>>>(att, BB * CC * CC);
  // q path: LN(cur; ln2) -> conv1x1(q_w1)
  ln_gemm_kernel<<<dim3(HWN / 64, 1, BB), 256, 0, stream>>>(cur, ln2w, ln2b, qw1, qb1, q1, CC);
  // kv path: LN(pre; ln1) -> conv1x1(kv_w1)
  ln_gemm_kernel<<<dim3(HWN / 64, 2, BB), 256, 0, stream>>>(pre, ln1w, ln1b, kvw1, kvb1, kv1, 2 * CC);
  dwconv_kernel<<<dim3((unsigned)((size_t)BB * CC * HWN / 256)), 256, 0, stream>>>(q1, qw2, qb2, q2, CC);
  dwconv_kernel<<<dim3((unsigned)((size_t)BB * 2 * CC * HWN / 256)), 256, 0, stream>>>(kv1, kvw2, kvb2, kv2, 2 * CC);
  l2norm_kernel<<<dim3(CC, BB, 2), 256, 0, stream>>>(q2, kv2, rq, rk);
  qk_gemm_kernel<<<dim3(HWN / 128, BB), 256, 0, stream>>>(q2, kv2, att);
  softmax_kernel<<<dim3(CC, BB), 128, 0, stream>>>(att, rq, rk, attH);
  av_out_kernel<<<dim3(HWN / 128, BB), 256, 0, stream>>>(attH, kv2, ow, ob, cur, out);
}
